// MambaModel_76639396430348
// MI455X (gfx1250) — compile-verified, matrix operand order checked by reading, not run on hardware
//
#include <hip/hip_runtime.h>
#include <hip/hip_bf16.h>

// ---------------------------------------------------------------------------
// Model constants (from the reference)
// ---------------------------------------------------------------------------
#define VOCAB 50288
#define DMODEL 768
#define NLAYER 24
#define EDIM 1536
#define NSTATE 128
#define PDIM 64
#define NHEAD 24
#define KCONV 4
#define CDIM 1792       // E + 2*N
#define INDIM 3352      // 2*E + 2*N + H
#define BATCH 8
#define SEQ 512
#define NTOK 4096       // BATCH*SEQ

typedef __attribute__((ext_vector_type(16))) __bf16 v16bf;
typedef __attribute__((ext_vector_type(8)))  __bf16 v8bf;
typedef __attribute__((ext_vector_type(4)))  __bf16 v4bf;
typedef __attribute__((ext_vector_type(8)))  float  v8f;
typedef __attribute__((ext_vector_type(4)))  float  v4f;

// f32 -> bf16 round-to-nearest-even
__device__ inline __bf16 f2bf(float f) {
  union { float f; unsigned u; } in; in.f = f;
  unsigned r = in.u + 0x7FFFu + ((in.u >> 16) & 1u);
  union { unsigned short u; __bf16 b; } out; out.u = (unsigned short)(r >> 16);
  return out.b;
}

__device__ inline v4bf f2bf4(v4f v) {
  v4bf r;
  r[0] = f2bf(v[0]); r[1] = f2bf(v[1]); r[2] = f2bf(v[2]); r[3] = f2bf(v[3]);
  return r;
}

__device__ inline float block_reduce_sum(float v, float* s) {
  int tid = threadIdx.x;
  s[tid] = v; __syncthreads();
  for (int off = 128; off > 0; off >>= 1) {
    if (tid < off) s[tid] += s[tid + off];
    __syncthreads();
  }
  float r = s[0]; __syncthreads();
  return r;
}

__device__ inline float silu(float x) { return x / (1.f + __expf(-x)); }

// ---------------------------------------------------------------------------
// Embedding gather + mask:  h[token][d] = emb[ids[token]][d] * mask[token]
// ---------------------------------------------------------------------------
__global__ __launch_bounds__(256) void embed_kernel(
    const int* __restrict__ ids, const float* __restrict__ mask,
    const float* __restrict__ emb, float* __restrict__ h) {
  long token = blockIdx.x;
  long id = ids[token];
  float m = mask[token];
  const float* er = emb + id * DMODEL;
  float* hr = h + token * DMODEL;
  #pragma unroll
  for (int j = 0; j < 3; ++j) {
    int d = threadIdx.x + j * 256;
    hr[d] = er[d] * m;
  }
}

// ---------------------------------------------------------------------------
// RMSNorm over 768 (used for layer input and final norm of last tokens)
// xr = x + blockIdx.x * xstride, out row stride is 768.
// ---------------------------------------------------------------------------
__global__ __launch_bounds__(256) void rmsnorm768_kernel(
    const float* __restrict__ x, const float* __restrict__ w,
    float* __restrict__ out, long xstride) {
  __shared__ float sred[256];
  const float* xr = x + (long)blockIdx.x * xstride;
  float* orow = out + (long)blockIdx.x * DMODEL;
  int tid = threadIdx.x;
  float v0 = xr[tid], v1 = xr[tid + 256], v2 = xr[tid + 512];
  float ss = block_reduce_sum(v0*v0 + v1*v1 + v2*v2, sred);
  float scale = rsqrtf(ss / (float)DMODEL + 1e-5f);
  orow[tid]       = v0 * scale * w[tid];
  orow[tid + 256] = v1 * scale * w[tid + 256];
  orow[tid + 512] = v2 * scale * w[tid + 512];
}

// ---------------------------------------------------------------------------
// WMMA GEMM:  C(MxN) = A(MxK) * B(KxN)  [+ Cin residual], all f32 in memory,
// bf16 v_wmma_f32_16x16x32_bf16 compute with f32 accumulation.
// Block: 256 threads (8 waves). Block tile 128x64x32. Wave tile 32x32 (2x2 WMMA).
// Staging: all 6 global b128 loads issued first (in flight together), then
// converted to bf16 and stored to LDS (A packed b64, B transposed b16).
// Requires Nn % 4 == 0, K % 32 == 0, M % 128 == 0 (true for all call sites).
// ---------------------------------------------------------------------------
#define BM 128
#define BN 64
#define BK 32
#define LPAD 40   // padded LDS row stride (bf16 elems): 80B, conflict-free b128 frags

__global__ __launch_bounds__(256) void gemm_bf16_wmma(
    const float* __restrict__ A, const float* __restrict__ B,
    const float* __restrict__ Cin, float* __restrict__ C,
    int M, int Nn, int Kd, int lda, int ldb, int ldc) {
  __shared__ __align__(16) __bf16 As[BM * LPAD];
  __shared__ __align__(16) __bf16 Bs[BN * LPAD];

  const int tid  = threadIdx.x;
  const int lane = tid & 31;
  const int wave = tid >> 5;
  const int wm   = wave >> 1;        // 0..3  (row group of 32)
  const int wn   = wave & 1;         // 0..1  (col group of 32)
  const int row0 = blockIdx.y * BM;
  const int col0 = blockIdx.x * BN;

  const v8f vzero = {0.f,0.f,0.f,0.f,0.f,0.f,0.f,0.f};
  v8f acc[2][2];
  acc[0][0] = vzero; acc[0][1] = vzero; acc[1][0] = vzero; acc[1][1] = vzero;

  // Per-thread staging coordinates (constant across K steps)
  int ar[4], ac4[4];
  #pragma unroll
  for (int i = 0; i < 4; ++i) {
    int q = tid + i * 256;           // 1024 quads = 128 rows * 8 quads
    ar[i]  = q >> 3;                 // row 0..127
    ac4[i] = (q & 7) << 2;           // col 0,4,...,28
  }
  int br[2], bc4[2], bok[2];
  #pragma unroll
  for (int i = 0; i < 2; ++i) {
    int q = tid + i * 256;           // 512 quads = 32 k-rows * 16 col-quads
    br[i]  = q >> 4;                 // k row 0..31
    bc4[i] = (q & 15) << 2;          // n col 0,4,...,60
    bok[i] = (col0 + bc4[i] < Nn);   // Nn % 4 == 0: quad all-in or all-out
  }

  // Fragment layout per CDNA5 ISA (16-bit 16x32 A / 32x16 B):
  //  lane<16 : M(or N)=lane,    elems 0..7 = K0..7,  elems 8..15 = K16..23
  //  lane>=16: M(or N)=lane-16, elems 0..7 = K8..15, elems 8..15 = K24..31
  const int frag_mn  = lane & 15;
  const int khalf    = (lane >> 4) << 3;   // 0 or 8 (bf16 elems)

  for (int k0 = 0; k0 < Kd; k0 += BK) {
    // ---- issue ALL staging loads first so they overlap in flight ----
    v4f aq[4];
    #pragma unroll
    for (int i = 0; i < 4; ++i)
      aq[i] = *reinterpret_cast<const v4f*>(
          &A[(long)(row0 + ar[i]) * lda + k0 + ac4[i]]);
    v4f bq[2];
    #pragma unroll
    for (int i = 0; i < 2; ++i) {
      v4f v = {0.f, 0.f, 0.f, 0.f};
      if (bok[i])
        v = *reinterpret_cast<const v4f*>(
            &B[(long)(k0 + br[i]) * ldb + col0 + bc4[i]]);
      bq[i] = v;
    }
    // prefetch next K tile (global_prefetch_b8)
    if (k0 + BK < Kd) {
      __builtin_prefetch(A + (long)(row0 + (tid >> 1)) * lda + k0 + BK + ((tid & 1) << 4), 0, 0);
      int pc = col0 + (tid & 63);
      if (pc < Nn)
        __builtin_prefetch(B + (long)(k0 + BK + (tid >> 6) * 8) * ldb + pc, 0, 0);
    }
    // ---- convert + store to LDS ----
    #pragma unroll
    for (int i = 0; i < 4; ++i)
      *reinterpret_cast<v4bf*>(&As[ar[i] * LPAD + ac4[i]]) = f2bf4(aq[i]);
    #pragma unroll
    for (int i = 0; i < 2; ++i)
      #pragma unroll
      for (int j = 0; j < 4; ++j)
        Bs[(bc4[i] + j) * LPAD + br[i]] = f2bf(bq[i][j]);
    __syncthreads();

    v16bf afrag[2], bfrag[2];
    #pragma unroll
    for (int tm = 0; tm < 2; ++tm) {
      int ro = wm * 32 + tm * 16 + frag_mn;
      v8bf lo = *reinterpret_cast<const v8bf*>(&As[ro * LPAD + khalf]);
      v8bf hi = *reinterpret_cast<const v8bf*>(&As[ro * LPAD + khalf + 16]);
      afrag[tm] = __builtin_shufflevector(lo, hi, 0,1,2,3,4,5,6,7,8,9,10,11,12,13,14,15);
    }
    #pragma unroll
    for (int tn = 0; tn < 2; ++tn) {
      int no = wn * 32 + tn * 16 + frag_mn;
      v8bf lo = *reinterpret_cast<const v8bf*>(&Bs[no * LPAD + khalf]);
      v8bf hi = *reinterpret_cast<const v8bf*>(&Bs[no * LPAD + khalf + 16]);
      bfrag[tn] = __builtin_shufflevector(lo, hi, 0,1,2,3,4,5,6,7,8,9,10,11,12,13,14,15);
    }
    #pragma unroll
    for (int tm = 0; tm < 2; ++tm)
      #pragma unroll
      for (int tn = 0; tn < 2; ++tn)
        acc[tm][tn] = __builtin_amdgcn_wmma_f32_16x16x32_bf16(
            false, afrag[tm], false, bfrag[tn], (short)0, acc[tm][tn], false, false);
    __syncthreads();
  }

  // ---- epilogue: C/D layout: lane<16 rows 0..7, lane>=16 rows 8..15; col=lane&15
  const int crow_off = (lane >> 4) << 3;
  const int ccol     = lane & 15;
  #pragma unroll
  for (int tm = 0; tm < 2; ++tm) {
    #pragma unroll
    for (int tn = 0; tn < 2; ++tn) {
      int gcol = col0 + wn * 32 + tn * 16 + ccol;
      if (gcol < Nn) {
        #pragma unroll
        for (int i = 0; i < 8; ++i) {
          long grow = row0 + wm * 32 + tm * 16 + crow_off + i;
          long idx = grow * ldc + gcol;
          float v = acc[tm][tn][i];
          C[idx] = Cin ? (Cin[idx] + v) : v;
        }
      }
    }
  }
}

// ---------------------------------------------------------------------------
// dt = softplus(zxbcdt[:, E+CD:] + dt_bias)
// ---------------------------------------------------------------------------
__global__ __launch_bounds__(256) void dt_kernel(
    const float* __restrict__ zx, const float* __restrict__ dtb,
    float* __restrict__ dt) {
  long idx = (long)blockIdx.x * 256 + threadIdx.x;
  if (idx >= (long)NTOK * NHEAD) return;
  long token = idx / NHEAD;
  int hh = (int)(idx % NHEAD);
  float x = zx[token * INDIM + (EDIM + CDIM) + hh] + dtb[hh];
  dt[token * NHEAD + hh] = (x > 20.f) ? x : log1pf(__expf(x));
}

// ---------------------------------------------------------------------------
// Depthwise causal conv (K=4) over sequence + bias + SiLU.
// Input xBC = zx[:, E : E+CD]; output xbc (NTOK x CD).
// ---------------------------------------------------------------------------
__global__ __launch_bounds__(256) void conv_silu_kernel(
    const float* __restrict__ zx, const float* __restrict__ w,
    const float* __restrict__ bias, float* __restrict__ out) {
  long idx = (long)blockIdx.x * 256 + threadIdx.x;
  if (idx >= (long)NTOK * CDIM) return;
  int c = (int)(idx % CDIM);
  long token = idx / CDIM;
  int s = (int)(token & (SEQ - 1));
  const float* wc = w + (long)c * KCONV;
  float acc = bias[c];
  #pragma unroll
  for (int k = 0; k < KCONV; ++k) {
    int sp = s - (KCONV - 1) + k;
    if (sp >= 0)
      acc += wc[k] * zx[(token - (KCONV - 1) + k) * INDIM + EDIM + c];
  }
  out[token * CDIM + c] = silu(acc);
}

// ---------------------------------------------------------------------------
// Selective SSM scan. One workgroup per (b,h); 256 threads.
// Thread (p = tid&63, ng = tid>>6) owns state[p][ng*32 .. ng*32+31] in regs.
// ---------------------------------------------------------------------------
__global__ __launch_bounds__(256) void ssm_scan_kernel(
    const float* __restrict__ xbc, const float* __restrict__ dt,
    const float* __restrict__ A_log, const float* __restrict__ Dp,
    float* __restrict__ y) {
  __shared__ float Bs[NSTATE];
  __shared__ float Cs[NSTATE];
  __shared__ float red[256];
  int h = blockIdx.x % NHEAD;
  int b = blockIdx.x / NHEAD;
  int tid = threadIdx.x;
  int p  = tid & 63;
  int ng = tid >> 6;
  const int nb = ng * 32;
  float Ah = -__expf(A_log[h]);
  float Dh = Dp[h];
  float st[32];
  #pragma unroll
  for (int j = 0; j < 32; ++j) st[j] = 0.f;

  for (int s = 0; s < SEQ; ++s) {
    long token = (long)b * SEQ + s;
    const float* xr = xbc + token * CDIM;
    if (tid < NSTATE) Bs[tid] = xr[EDIM + tid];
    else if (tid < 2 * NSTATE) Cs[tid - NSTATE] = xr[EDIM + NSTATE + (tid - NSTATE)];
    __syncthreads();
    float dtv = dt[token * NHEAD + h];
    float dAv = __expf(dtv * Ah);
    float xv  = xr[h * PDIM + p];
    float dtx = dtv * xv;
    float acc = 0.f;
    #pragma unroll
    for (int j = 0; j < 32; ++j) {
      st[j] = st[j] * dAv + dtx * Bs[nb + j];
      acc += st[j] * Cs[nb + j];
    }
    red[tid] = acc;
    __syncthreads();
    if (ng == 0)
      y[token * EDIM + h * PDIM + p] =
          red[p] + red[64 + p] + red[128 + p] + red[192 + p] + Dh * xv;
    __syncthreads();
  }
}

// ---------------------------------------------------------------------------
// yg = rmsnorm(y * silu(z), gn_w)  over E=1536 per token
// ---------------------------------------------------------------------------
__global__ __launch_bounds__(256) void gate_norm_kernel(
    const float* __restrict__ y, const float* __restrict__ zx,
    const float* __restrict__ gnw, float* __restrict__ yg) {
  __shared__ float sred[256];
  long token = blockIdx.x;
  const float* yr = y + token * EDIM;
  const float* zr = zx + token * INDIM;
  float vals[6];
  float ss = 0.f;
  #pragma unroll
  for (int j = 0; j < 6; ++j) {
    int e = threadIdx.x + j * 256;
    float g = yr[e] * silu(zr[e]);
    vals[j] = g;
    ss += g * g;
  }
  ss = block_reduce_sum(ss, sred);
  float scale = rsqrtf(ss / (float)EDIM + 1e-5f);
  float* o = yg + token * EDIM;
  #pragma unroll
  for (int j = 0; j < 6; ++j) {
    int e = threadIdx.x + j * 256;
    o[e] = vals[j] * scale * gnw[e];
  }
}

// ---------------------------------------------------------------------------
// logits[b][v] = dot(hf[b], emb[v]) — bandwidth bound; 8 vocab rows per block,
// one wave per row, 8 hidden vectors cached in LDS.
// ---------------------------------------------------------------------------
__global__ __launch_bounds__(256) void logits_kernel(
    const float* __restrict__ hf, const float* __restrict__ emb,
    float* __restrict__ logits) {
  __shared__ float sh[BATCH * DMODEL];
  int tid = threadIdx.x;
  #pragma unroll
  for (int i = 0; i < 24; ++i) sh[tid + i * 256] = hf[tid + i * 256];
  __syncthreads();
  int v = blockIdx.x * 8 + (tid >> 5);
  int lane = tid & 31;
  if (v < VOCAB) {
    const float* er = emb + (long)v * DMODEL;
    float acc[BATCH];
    #pragma unroll
    for (int bb = 0; bb < BATCH; ++bb) acc[bb] = 0.f;
    for (int d = lane; d < DMODEL; d += 32) {
      float e = er[d];
      #pragma unroll
      for (int bb = 0; bb < BATCH; ++bb) acc[bb] += e * sh[bb * DMODEL + d];
    }
    #pragma unroll
    for (int bb = 0; bb < BATCH; ++bb) {
      float a = acc[bb];
      for (int off = 16; off > 0; off >>= 1) a += __shfl_down(a, off);
      if (lane == 0) logits[(long)bb * VOCAB + v] = a;
    }
  }
}

// ---------------------------------------------------------------------------
// Per-batch -log_softmax at label
// ---------------------------------------------------------------------------
__global__ __launch_bounds__(256) void loss_kernel(
    const float* __restrict__ logits, const int* __restrict__ labels,
    float* __restrict__ lossp) {
  __shared__ float sred[256];
  int b = blockIdx.x, tid = threadIdx.x;
  const float* lr = logits + (long)b * VOCAB;
  float m = -1e30f;
  for (int i = tid; i < VOCAB; i += 256) m = fmaxf(m, lr[i]);
  sred[tid] = m; __syncthreads();
  for (int off = 128; off > 0; off >>= 1) {
    if (tid < off) sred[tid] = fmaxf(sred[tid], sred[tid + off]);
    __syncthreads();
  }
  m = sred[0]; __syncthreads();
  float s = 0.f;
  for (int i = tid; i < VOCAB; i += 256) s += __expf(lr[i] - m);
  s = block_reduce_sum(s, sred);
  if (tid == 0) lossp[b] = -(lr[labels[b]] - m - logf(s));
}

__global__ void finalize_kernel(const float* __restrict__ lossp,
                                const float* __restrict__ logits,
                                float* __restrict__ out) {
  int tid = threadIdx.x;
  if (tid == 0) {
    float s = 0.f;
    for (int b = 0; b < BATCH; ++b) s += lossp[b];
    out[0] = s / (float)BATCH;
  }
  if (tid >= 1 && tid < 33) {
    const int TOK[4] = {317, 347, 327, 360};
    int j = tid - 1;
    out[1 + j] = logits[(long)(j >> 2) * VOCAB + TOK[j & 3]];
  }
}

// ---------------------------------------------------------------------------
// Host launcher
// ---------------------------------------------------------------------------
extern "C" void kernel_launch(void* const* d_in, const int* in_sizes, int n_in,
                              void* d_out, int out_size, void* d_ws, size_t ws_size,
                              hipStream_t stream) {
  const int*   ids     = (const int*)d_in[0];
  const float* mask    = (const float*)d_in[1];
  const int*   labels  = (const int*)d_in[2];
  const float* emb     = (const float*)d_in[3];
  const float* ln_w    = (const float*)d_in[4];
  const float* in_proj = (const float*)d_in[5];
  const float* conv_w  = (const float*)d_in[6];
  const float* conv_b  = (const float*)d_in[7];
  const float* dt_bias = (const float*)d_in[8];
  const float* A_log   = (const float*)d_in[9];
  const float* Dpar    = (const float*)d_in[10];
  const float* gn_w    = (const float*)d_in[11];
  const float* outproj = (const float*)d_in[12];
  const float* norm_f  = (const float*)d_in[13];

  float* ws = (float*)d_ws;
  float* h      = ws;                        // 4096*768
  float* xn     = h      + (long)NTOK * DMODEL;
  float* zx     = xn     + (long)NTOK * DMODEL;      // 4096*3352
  float* dtb    = zx     + (long)NTOK * INDIM;       // 4096*24
  float* xbc    = dtb    + (long)NTOK * NHEAD;       // 4096*1792
  float* y      = xbc    + (long)NTOK * CDIM;        // 4096*1536
  float* yg     = y      + (long)NTOK * EDIM;        // 4096*1536
  float* hf     = yg     + (long)NTOK * EDIM;        // 8*768
  float* logits = hf     + (long)BATCH * DMODEL;     // 8*50288
  float* lossp  = logits + (long)BATCH * VOCAB;      // 8

  embed_kernel<<<NTOK, 256, 0, stream>>>(ids, mask, emb, h);

  for (int l = 0; l < NLAYER; ++l) {
    rmsnorm768_kernel<<<NTOK, 256, 0, stream>>>(h, ln_w + (long)l * DMODEL, xn, DMODEL);

    // zxbcdt = xn @ in_proj[l]   (4096x768 * 768x3352)
    gemm_bf16_wmma<<<dim3((INDIM + BN - 1) / BN, NTOK / BM), 256, 0, stream>>>(
        xn, in_proj + (long)l * DMODEL * INDIM, nullptr, zx,
        NTOK, INDIM, DMODEL, DMODEL, INDIM, INDIM);

    dt_kernel<<<(NTOK * NHEAD + 255) / 256, 256, 0, stream>>>(
        zx, dt_bias + (long)l * NHEAD, dtb);

    conv_silu_kernel<<<(int)(((long)NTOK * CDIM + 255) / 256), 256, 0, stream>>>(
        zx, conv_w + (long)l * CDIM * KCONV, conv_b + (long)l * CDIM, xbc);

    ssm_scan_kernel<<<BATCH * NHEAD, 256, 0, stream>>>(
        xbc, dtb, A_log + (long)l * NHEAD, Dpar + (long)l * NHEAD, y);

    gate_norm_kernel<<<NTOK, 256, 0, stream>>>(y, zx, gn_w + (long)l * EDIM, yg);

    // h = h + yg @ out_proj[l]   (4096x1536 * 1536x768, residual)
    gemm_bf16_wmma<<<dim3(DMODEL / BN, NTOK / BM), 256, 0, stream>>>(
        yg, outproj + (long)l * EDIM * DMODEL, h, h,
        NTOK, DMODEL, EDIM, EDIM, DMODEL, DMODEL);
  }

  // final RMSNorm on the 8 last-token rows only
  rmsnorm768_kernel<<<BATCH, 256, 0, stream>>>(
      h + (long)(SEQ - 1) * DMODEL, norm_f, hf, (long)SEQ * DMODEL);

  logits_kernel<<<(VOCAB + 7) / 8, 256, 0, stream>>>(hf, emb, logits);
  loss_kernel<<<BATCH, 256, 0, stream>>>(logits, labels, lossp);
  finalize_kernel<<<1, 64, 0, stream>>>(lossp, logits, (float*)d_out);
}
